// RingAttention_37641093382542
// MI455X (gfx1250) — matrix operand order repaired and checked by reading, not compiled, not run on hardware
//
#include <hip/hip_runtime.h>

typedef __attribute__((ext_vector_type(16))) __bf16 v16bf;
typedef __attribute__((ext_vector_type(8)))  __bf16 v8bf;
typedef __attribute__((ext_vector_type(4)))  __bf16 v4bf;
typedef __attribute__((ext_vector_type(8)))  float  v8f;
typedef __attribute__((ext_vector_type(4)))  float  v4f;
typedef __attribute__((ext_vector_type(4)))  unsigned int u32x4;
typedef __attribute__((ext_vector_type(8)))  int    i32x8;
typedef __attribute__((ext_vector_type(4)))  int    i32x4;

#define NSEQ   2048
#define DMODEL 1024
#define NHEAD  16
#define DK     64
#define KVBS   256

#if __has_builtin(__builtin_amdgcn_tensor_load_to_lds) && \
    __has_builtin(__builtin_amdgcn_s_wait_tensorcnt)
#define HAVE_TDM 1
#else
#define HAVE_TDM 0
#endif

__device__ __forceinline__ v8f vzero8() {
  v8f z;
#pragma unroll
  for (int j = 0; j < 8; ++j) z[j] = 0.0f;
  return z;
}

__device__ __forceinline__ v16bf cat16(v8bf a, v8bf b) {
  v16bf r;
#pragma unroll
  for (int j = 0; j < 8; ++j) { r[j] = a[j]; r[8 + j] = b[j]; }
  return r;
}

// A fragment (16x32 bf16). smem row-major [row][k], stride in elements.
__device__ __forceinline__ v16bf load_a(const __bf16* s, int stride, int m0, int kb, int lane) {
  int row = m0 + (lane & 15);
  int kh  = (lane & 16) ? 8 : 0;
  const __bf16* p = s + row * stride + kb + kh;
  v8bf c0 = *(const v8bf*)(p);
  v8bf c1 = *(const v8bf*)(p + 16);
  return cat16(c0, c1);
}

// B fragment (32x16 bf16). smem stores B^T: [n][k] rows, k contiguous.
__device__ __forceinline__ v16bf load_b(const __bf16* s, int stride, int n0, int kb, int lane) {
  int n  = n0 + (lane & 15);
  int kh = (lane & 16) ? 8 : 0;
  const __bf16* p = s + n * stride + kb + kh;
  v8bf c0 = *(const v8bf*)(p);
  v8bf c1 = *(const v8bf*)(p + 16);
  return cat16(c0, c1);
}

__device__ __forceinline__ v8f wmma_bf16(v16bf a, v16bf b, v8f c) {
  return __builtin_amdgcn_wmma_f32_16x16x32_bf16(false, a, false, b, (short)0, c, false, false);
}

// ---------------------------------------------------------------------------
// TDM: async 2D tile load, bf16, 64-elem rows (128B = 32 DWORDs), LDS rows
// padded by 4 DWORDs -> element stride 72.  D# per ISA cdna5 ch.8.
// ---------------------------------------------------------------------------
__device__ __forceinline__ void tdm_load_tile64(unsigned lds_off, const void* gptr,
                                                int rows) {
#if HAVE_TDM
  unsigned long long ga = (unsigned long long)(uintptr_t)gptr;
  u32x4 g0;
  g0[0] = 1u;                                          // count=1 (valid user D#)
  g0[1] = lds_off;                                     // lds_addr (bytes)
  g0[2] = (unsigned)(ga & 0xFFFFFFFFull);              // global_addr[31:0]
  g0[3] = (unsigned)((ga >> 32) & 0x1FFFFFFull)        // global_addr[56:32]
          | (2u << 30);                                // type = 2 ("image")
  i32x8 g1;
  g1[0] = (1 << 16)     // data_size = 1 -> 2 bytes
        | (1 << 20)     // pad_enable
        | (4 << 22)     // pad_interval: code 4 -> every 32 DWORDs
        | (3 << 25);    // pad_amount:   code 3 -> 4 DWORDs (8 bf16)
  g1[1] = (int)(64u << 16);                 // tensor_dim0 = 64 (bits 79:48)
  g1[2] = (int)((unsigned)rows << 16);      // tensor_dim1 = rows (bits 111:80)
  g1[3] = (int)(64u << 16);                 // tile_dim0 = 64 (bits 127:112)
  g1[4] = rows;                             // tile_dim1 = rows; tile_dim2 = 0
  g1[5] = 64;                               // tensor_dim0_stride = 64
  g1[6] = 0;
  g1[7] = 0;
  i32x4 gz = {0, 0, 0, 0};
#if defined(__clang_major__) && (__clang_major__ >= 23)
  i32x8 g4 = {0, 0, 0, 0, 0, 0, 0, 0};
  __builtin_amdgcn_tensor_load_to_lds(g0, g1, gz, gz, g4, 0);
#else
  __builtin_amdgcn_tensor_load_to_lds(g0, g1, gz, gz, 0);
#endif
#else
  (void)lds_off; (void)gptr; (void)rows;
#endif
}

__device__ __forceinline__ void tdm_wait0() {
#if HAVE_TDM
  __builtin_amdgcn_s_wait_tensorcnt(0);
#endif
}

__device__ __forceinline__ unsigned lds_offset_of(const void* p) {
  return (unsigned)(uintptr_t)p;   // generic LDS ptr: addr[31:0] = LDS offset
}

// ---------------------------------------------------------------------------
// Kernel 1: fused QKV projection. grid = (16, 8, 3), block = 256 (8 waves).
// ---------------------------------------------------------------------------
__global__ __launch_bounds__(256) void qkv_proj_kernel(
    const float* __restrict__ x,
    const float* __restrict__ Wq, const float* __restrict__ bq,
    const float* __restrict__ Wk, const float* __restrict__ bk,
    const float* __restrict__ Wv, const float* __restrict__ bv,
    __bf16* __restrict__ Qw, __bf16* __restrict__ Kw, __bf16* __restrict__ Vw)
{
  __shared__ __bf16 As[128 * 72];
  __shared__ __bf16 Bs[128 * 72];

  const int tid  = threadIdx.x;
  const int lane = tid & 31;
  const int wave = tid >> 5;
  const int m0w  = (wave & 3) * 32;
  const int n0w  = (wave >> 2) * 64;
  const int mb   = blockIdx.x * 128;
  const int ob   = blockIdx.y * 128;

  const float* W; const float* bias; __bf16* dst;
  if (blockIdx.z == 0)      { W = Wq; bias = bq; dst = Qw; }
  else if (blockIdx.z == 1) { W = Wk; bias = bk; dst = Kw; }
  else                      { W = Wv; bias = bv; dst = Vw; }

  v8f acc[2][4];
#pragma unroll
  for (int i = 0; i < 2; ++i)
#pragma unroll
    for (int j = 0; j < 4; ++j) acc[i][j] = vzero8();

  for (int kc = 0; kc < DMODEL; kc += 64) {
    __syncthreads();
#pragma unroll
    for (int it = 0; it < 8; ++it) {
      int c   = tid + it * 256;
      int row = c >> 4;
      int c4  = (c & 15) << 2;
      v4f a4 = *(const v4f*)(x + (size_t)(mb + row) * DMODEL + kc + c4);
      v4f b4 = *(const v4f*)(W + (size_t)(ob + row) * DMODEL + kc + c4);
      v4bf ab, bb;
#pragma unroll
      for (int j = 0; j < 4; ++j) { ab[j] = (__bf16)a4[j]; bb[j] = (__bf16)b4[j]; }
      *(v4bf*)(As + row * 72 + c4) = ab;
      *(v4bf*)(Bs + row * 72 + c4) = bb;
    }
    __syncthreads();
#pragma unroll
    for (int ks = 0; ks < 64; ks += 32) {
      v16bf a[2], b[4];
      a[0] = load_a(As, 72, m0w,      ks, lane);
      a[1] = load_a(As, 72, m0w + 16, ks, lane);
#pragma unroll
      for (int nt = 0; nt < 4; ++nt) b[nt] = load_b(Bs, 72, n0w + nt * 16, ks, lane);
#pragma unroll
      for (int mt = 0; mt < 2; ++mt)
#pragma unroll
        for (int nt = 0; nt < 4; ++nt)
          acc[mt][nt] = wmma_bf16(a[mt], b[nt], acc[mt][nt]);
    }
  }

  const int rbase = mb + m0w + ((lane & 16) ? 8 : 0);
#pragma unroll
  for (int nt = 0; nt < 4; ++nt) {
    int o = ob + n0w + nt * 16 + (lane & 15);
    float bv0 = bias[o];
    int h = o >> 6, d = o & 63;
#pragma unroll
    for (int mt = 0; mt < 2; ++mt)
#pragma unroll
      for (int i = 0; i < 8; ++i) {
        int row = rbase + mt * 16 + i;
        dst[((size_t)h * NSEQ + row) * DK + d] = (__bf16)(acc[mt][nt][i] + bv0);
      }
  }
}

// ---------------------------------------------------------------------------
// Kernel 2: fused blocked-softmax attention. grid = (16, 16), block = 256.
// Q/K tiles arrive via TDM (tensor_load_to_lds, padded rows); V^T staged
// manually (TDM has no transpose).
// ---------------------------------------------------------------------------
__global__ __launch_bounds__(256) void blocksoftmax_attn_kernel(
    const __bf16* __restrict__ Qw, const __bf16* __restrict__ Kw,
    const __bf16* __restrict__ Vw, const float* __restrict__ bias,
    __bf16* __restrict__ Ow)
{
  __shared__ __bf16 Qs[128 * 72];        // [qrow][d]
  __shared__ __bf16 Ks[256 * 72];        // [key][d]
  __shared__ __bf16 Vt[64 * 264];        // [d][key]
  __shared__ __bf16 Ps[8][16 * 264];     // per-wave weights [qrow][key]

  const int tid  = threadIdx.x;
  const int lane = tid & 31;
  const int wave = tid >> 5;
  const int h    = blockIdx.y;
  const int qb0  = blockIdx.x * 128;
  const int hh   = (lane & 16) ? 8 : 0;
  const int q0w  = wave * 16;
  const float scale = 0.125f;            // 1/sqrt(64)

  // ---- stage Q tile (TDM async, else manual) ----
  if (HAVE_TDM) {
    if (wave == 0)
      tdm_load_tile64(lds_offset_of(Qs),
                      Qw + ((size_t)h * NSEQ + qb0) * DK, 128);
  } else {
#pragma unroll
    for (int it = 0; it < 4; ++it) {
      int c = tid + it * 256;
      int row = c >> 3, db = (c & 7) << 3;
      *(v8bf*)(Qs + row * 72 + db) =
          *(const v8bf*)(Qw + ((size_t)h * NSEQ + qb0 + row) * DK + db);
    }
  }

  v8f oacc[4];
#pragma unroll
  for (int j = 0; j < 4; ++j) oacc[j] = vzero8();

  __bf16* pw = &Ps[wave][0];

  for (int kb = 0; kb < 8; ++kb) {
    __syncthreads();
    // ---- K tile via TDM (async DMA into padded LDS rows) ----
    if (HAVE_TDM && wave == 0)
      tdm_load_tile64(lds_offset_of(Ks),
                      Kw + ((size_t)h * NSEQ + kb * KVBS) * DK, 256);
    if (kb < 7)
      __builtin_prefetch(Vw + ((size_t)h * NSEQ + (kb + 1) * KVBS) * DK + tid * 8, 0, 1);
    // ---- V^T staging (manual transpose); K manual only without TDM ----
#pragma unroll
    for (int it = 0; it < 8; ++it) {
      int c = tid + it * 256;
      int key = c >> 3, db = (c & 7) << 3;
      if (!HAVE_TDM)
        *(v8bf*)(Ks + key * 72 + db) =
            *(const v8bf*)(Kw + ((size_t)h * NSEQ + kb * KVBS + key) * DK + db);
      v8bf v = *(const v8bf*)(Vw + ((size_t)h * NSEQ + kb * KVBS + key) * DK + db);
#pragma unroll
      for (int j = 0; j < 8; ++j) Vt[(db + j) * 264 + key] = v[j];
    }
    if (HAVE_TDM && wave == 0) tdm_wait0();   // Q (first iter) + K complete
    __syncthreads();

    // ---- S = Q K^T (16 q rows x 256 keys, fp32 accum) ----
    v16bf aq0 = load_a(Qs, 72, q0w, 0,  lane);
    v16bf aq1 = load_a(Qs, 72, q0w, 32, lane);
    v8f s[16];
#pragma unroll
    for (int t = 0; t < 16; ++t) {
      v8f a = vzero8();
      a = wmma_bf16(aq0, load_b(Ks, 72, t * 16, 0,  lane), a);
      a = wmma_bf16(aq1, load_b(Ks, 72, t * 16, 32, lane), a);
      s[t] = a;
    }

    // ---- scale - bias, per-block softmax over 256 keys ----
    float mrow[8];
#pragma unroll
    for (int i = 0; i < 8; ++i) mrow[i] = -3.0e38f;
#pragma unroll
    for (int t = 0; t < 16; ++t) {
      int kcol = kb * KVBS + t * 16 + (lane & 15);
#pragma unroll
      for (int i = 0; i < 8; ++i) {
        int qrow = qb0 + q0w + hh + i;
        float v = s[t][i] * scale - bias[(size_t)qrow * NSEQ + kcol];
        s[t][i] = v;
        mrow[i] = fmaxf(mrow[i], v);
      }
    }
#pragma unroll
    for (int off = 8; off >= 1; off >>= 1)
#pragma unroll
      for (int i = 0; i < 8; ++i)
        mrow[i] = fmaxf(mrow[i], __shfl_xor(mrow[i], off, 32));

    float rs[8];
#pragma unroll
    for (int i = 0; i < 8; ++i) rs[i] = 0.0f;
#pragma unroll
    for (int t = 0; t < 16; ++t)
#pragma unroll
      for (int i = 0; i < 8; ++i) {
        float p = __expf(s[t][i] - mrow[i]);
        s[t][i] = p;
        rs[i] += p;
      }
#pragma unroll
    for (int off = 8; off >= 1; off >>= 1)
#pragma unroll
      for (int i = 0; i < 8; ++i)
        rs[i] += __shfl_xor(rs[i], off, 32);

    float rinv[8];
#pragma unroll
    for (int i = 0; i < 8; ++i) rinv[i] = 1.0f / rs[i];

    // ---- C-layout -> A-layout via per-wave LDS strip ----
#pragma unroll
    for (int t = 0; t < 16; ++t)
#pragma unroll
      for (int i = 0; i < 8; ++i)
        pw[(hh + i) * 264 + t * 16 + (lane & 15)] = (__bf16)(s[t][i] * rinv[i]);

    // ---- O += W @ V_block ----
#pragma unroll
    for (int ks = 0; ks < 8; ++ks) {
      v16bf aw = load_a(pw, 264, 0, ks * 32, lane);
#pragma unroll
      for (int nt = 0; nt < 4; ++nt)
        oacc[nt] = wmma_bf16(aw, load_b(Vt, 264, nt * 16, ks * 32, lane), oacc[nt]);
    }
  }

  const float inv_norm = 1.0f / (8.0f + 1e-8f);   // norm = nb exactly
#pragma unroll
  for (int nt = 0; nt < 4; ++nt) {
    int d = nt * 16 + (lane & 15);
#pragma unroll
    for (int i = 0; i < 8; ++i) {
      int qrow = qb0 + q0w + hh + i;
      Ow[(size_t)qrow * DMODEL + h * DK + d] = (__bf16)(oacc[nt][i] * inv_norm);
    }
  }
}

// ---------------------------------------------------------------------------
// Kernel 3: output projection. grid = (16, 8), block = 256. fp32 out + bo.
// ---------------------------------------------------------------------------
__global__ __launch_bounds__(256) void out_proj_kernel(
    const __bf16* __restrict__ Ow, const float* __restrict__ Wo,
    const float* __restrict__ bo, float* __restrict__ out)
{
  __shared__ __bf16 As[128 * 72];
  __shared__ __bf16 Bs[128 * 72];

  const int tid  = threadIdx.x;
  const int lane = tid & 31;
  const int wave = tid >> 5;
  const int m0w  = (wave & 3) * 32;
  const int n0w  = (wave >> 2) * 64;
  const int mb   = blockIdx.x * 128;
  const int ob   = blockIdx.y * 128;

  v8f acc[2][4];
#pragma unroll
  for (int i = 0; i < 2; ++i)
#pragma unroll
    for (int j = 0; j < 4; ++j) acc[i][j] = vzero8();

  for (int kc = 0; kc < DMODEL; kc += 64) {
    __syncthreads();
#pragma unroll
    for (int it = 0; it < 4; ++it) {     // A: bf16 chunks straight from ws
      int c = tid + it * 256;
      int row = c >> 3, db = (c & 7) << 3;
      *(v8bf*)(As + row * 72 + db) =
          *(const v8bf*)(Ow + (size_t)(mb + row) * DMODEL + kc + db);
    }
#pragma unroll
    for (int it = 0; it < 8; ++it) {     // B: fp32 -> bf16 convert
      int c = tid + it * 256;
      int row = c >> 4, c4 = (c & 15) << 2;
      v4f b4 = *(const v4f*)(Wo + (size_t)(ob + row) * DMODEL + kc + c4);
      v4bf bb;
#pragma unroll
      for (int j = 0; j < 4; ++j) bb[j] = (__bf16)b4[j];
      *(v4bf*)(Bs + row * 72 + c4) = bb;
    }
    __syncthreads();
#pragma unroll
    for (int ks = 0; ks < 64; ks += 32) {
      v16bf a[2], b[4];
      a[0] = load_a(As, 72, m0w,      ks, lane);
      a[1] = load_a(As, 72, m0w + 16, ks, lane);
#pragma unroll
      for (int nt = 0; nt < 4; ++nt) b[nt] = load_b(Bs, 72, n0w + nt * 16, ks, lane);
#pragma unroll
      for (int mt = 0; mt < 2; ++mt)
#pragma unroll
        for (int nt = 0; nt < 4; ++nt)
          acc[mt][nt] = wmma_bf16(a[mt], b[nt], acc[mt][nt]);
    }
  }

  const int rbase = mb + m0w + ((lane & 16) ? 8 : 0);
#pragma unroll
  for (int nt = 0; nt < 4; ++nt) {
    int o = ob + n0w + nt * 16 + (lane & 15);
    float b0 = bo[o];
#pragma unroll
    for (int mt = 0; mt < 2; ++mt)
#pragma unroll
      for (int i = 0; i < 8; ++i) {
        int row = rbase + mt * 16 + i;
        out[(size_t)row * DMODEL + o] = acc[mt][nt][i] + b0;
      }
  }
}

// ---------------------------------------------------------------------------
extern "C" void kernel_launch(void* const* d_in, const int* in_sizes, int n_in,
                              void* d_out, int out_size, void* d_ws, size_t ws_size,
                              hipStream_t stream) {
  (void)in_sizes; (void)n_in; (void)out_size; (void)ws_size;
  const float* x    = (const float*)d_in[0];
  const float* bias = (const float*)d_in[1];
  const float* Wq   = (const float*)d_in[2];
  const float* bq   = (const float*)d_in[3];
  const float* Wk   = (const float*)d_in[4];
  const float* bk   = (const float*)d_in[5];
  const float* Wv   = (const float*)d_in[6];
  const float* bv   = (const float*)d_in[7];
  const float* Wo   = (const float*)d_in[8];
  const float* bo   = (const float*)d_in[9];
  float* out = (float*)d_out;

  const size_t hnd = (size_t)NHEAD * NSEQ * DK;
  __bf16* Qw = (__bf16*)d_ws;
  __bf16* Kw = Qw + hnd;
  __bf16* Vw = Kw + hnd;
  __bf16* Ow = Vw + hnd;

  qkv_proj_kernel<<<dim3(NSEQ / 128, DMODEL / 128, 3), 256, 0, stream>>>(
      x, Wq, bq, Wk, bk, Wv, bv, Qw, Kw, Vw);
  blocksoftmax_attn_kernel<<<dim3(NSEQ / 128, NHEAD), 256, 0, stream>>>(
      Qw, Kw, Vw, bias, Ow);
  out_proj_kernel<<<dim3(NSEQ / 128, DMODEL / 128), 256, 0, stream>>>(
      Ow, Wo, bo, out);
}
